// Model_62328565400233
// MI455X (gfx1250) — matrix operand order repaired and checked by reading, not compiled, not run on hardware
//
#include <hip/hip_runtime.h>
#include <hip/hip_bf16.h>
#include <math.h>

typedef _Float16 v16h __attribute__((ext_vector_type(16)));
typedef _Float16 v8h  __attribute__((ext_vector_type(8)));
typedef float    v8f  __attribute__((ext_vector_type(8)));

#define B_   200
#define T_   250
#define F_   14
#define H_   256
#define D_   164
#define C_   8
#define RP   208      /* padded batch rows = 13*16 */
#define NBLK 16       /* one block per H-column tile */
#define NTHR 416      /* 13 waves = 13 row tiles */
#define RPH  (RP * H_)

// ---------------- workspace layout (bytes) ----------------
static constexpr size_t OFF_BAR = 0;                              // 2 ints
static constexpr size_t OFF_BP0 = 256;                            // W1 padded, B-form [1024][32] f16
static constexpr size_t OFF_BP1 = OFF_BP0 + (size_t)1024*32*2;    // U1 B-form   [1024][256] f16
static constexpr size_t OFF_BP2 = OFF_BP1 + (size_t)1024*256*2;   // [W2;U2] B-form [1024][512] f16
static constexpr size_t OFF_XP  = OFF_BP2 + (size_t)1024*512*2;   // x padded [T][208][32] f16
static constexpr size_t OFF_H1  = OFF_XP  + (size_t)T_*RP*32*2;   // h1 x2, h2 x2 (f16, contiguous)
static constexpr size_t OFF_C1  = OFF_H1  + (size_t)4*RPH*2;      // c1 f32 [208][256]
static constexpr size_t OFF_C2  = OFF_C1  + (size_t)RPH*4;        // c2 f32
static constexpr size_t OFF_H2F = OFF_C2  + (size_t)RPH*4;        // final h2 f32
static constexpr size_t OFF_END = OFF_H2F + (size_t)RPH*4;

// LDS layout (halves): per-block weight slab for its 16 columns x 4 gates
#define LB0_OFF 0                      /* [g*16+nl][32]  -> 2048 halves  */
#define LB1_OFF 2048                   /* [g*16+nl][256] -> 16384 halves */
#define LB2_OFF (2048 + 16384)         /* [g*16+nl][512] -> 32768 halves */
#define LDS_HALVES (2048 + 16384 + 32768)
#define LDS_BYTES  (LDS_HALVES * 2)

// ---------------- WMMA helpers ----------------
// A tile (16x32 f16): lane L holds row M=L%16; K = (L>=16?16:0) + i
// -> one contiguous 32B run at kb+kh*16.
__device__ __forceinline__ v16h load_a(const _Float16* __restrict__ base, int stride,
                                       int row, int kb, int kh) {
  const _Float16* p = base + (size_t)row * stride + kb + kh * 16;
  v8h lo = *(const v8h*)(p);
  v8h hi = *(const v8h*)(p + 8);
  return __builtin_shufflevector(lo, hi, 0,1,2,3,4,5,6,7,8,9,10,11,12,13,14,15);
}

// B tile (32x16 f16) from LDS: lane L holds col N=L%16; K = (L>=16?16:0)+i,
// one contiguous 32B run -> two ds_load_b128.  Plain (non-volatile) loads;
// hoisting is blocked by the opaque per-step offset added by the caller.
__device__ __forceinline__ v16h lds_b(const _Float16* lds, int idx) {
  v8h lo = *(const v8h*)(lds + idx);
  v8h hi = *(const v8h*)(lds + idx + 8);
  return __builtin_shufflevector(lo, hi, 0,1,2,3,4,5,6,7,8,9,10,11,12,13,14,15);
}

__device__ __forceinline__ v8f wmma_f16(v16h a, v16h b, v8f c) {
  return __builtin_amdgcn_wmma_f32_16x16x32_f16(false, a, false, b, (short)0, c,
                                                false, false);
}

// Opaque zero: defeats LICM of LDS loads across the 250-step loop without
// changing addressing mode or scope of the loads themselves.
__device__ __forceinline__ int opaque_zero() {
  int z;
  asm volatile("s_mov_b32 %0, 0" : "=s"(z));
  return z;
}

__device__ __forceinline__ float sigf(float x) { return 1.0f / (1.0f + expf(-x)); }

// ---------------- software grid barrier (16 co-resident blocks) ----------------
__device__ __forceinline__ void grid_barrier(int* bar) {
  __threadfence();
  __syncthreads();
  if (threadIdx.x == 0) {
    int* cnt = bar;
    int* gen = bar + 1;
    int g = __hip_atomic_load(gen, __ATOMIC_RELAXED, __HIP_MEMORY_SCOPE_AGENT);
    int v = __hip_atomic_fetch_add(cnt, 1, __ATOMIC_ACQ_REL, __HIP_MEMORY_SCOPE_AGENT);
    if (v == NBLK - 1) {
      __hip_atomic_store(cnt, 0, __ATOMIC_RELAXED, __HIP_MEMORY_SCOPE_AGENT);
      __hip_atomic_fetch_add(gen, 1, __ATOMIC_ACQ_REL, __HIP_MEMORY_SCOPE_AGENT);
    } else {
      while (__hip_atomic_load(gen, __ATOMIC_ACQUIRE, __HIP_MEMORY_SCOPE_AGENT) == g)
        __builtin_amdgcn_s_sleep(1);
    }
  }
  __syncthreads();
}

// ---------------- setup kernels ----------------
__global__ void k_init(int* bar, float* state, int nwords) {
  int i = blockIdx.x * blockDim.x + threadIdx.x;
  if (i < 2) bar[i] = 0;
  int stride = gridDim.x * blockDim.x;
  for (int w = i; w < nwords; w += stride) state[w] = 0.0f;
}

__global__ void k_prep(const float* __restrict__ W1, const float* __restrict__ U1,
                       const float* __restrict__ W2, const float* __restrict__ U2,
                       const float* __restrict__ x,
                       _Float16* __restrict__ bp0, _Float16* __restrict__ bp1,
                       _Float16* __restrict__ bp2, _Float16* __restrict__ xp) {
  int i0 = blockIdx.x * blockDim.x + threadIdx.x;
  int stride = gridDim.x * blockDim.x;
  for (int i = i0; i < 1024 * 32; i += stride) {          // W1 pad -> [1024][32]
    int n = i >> 5, k = i & 31;
    bp0[i] = (_Float16)(k < F_ ? W1[k * 1024 + n] : 0.0f);
  }
  for (int i = i0; i < 1024 * 256; i += stride) {         // U1 -> [1024][256]
    int n = i >> 8, k = i & 255;
    bp1[i] = (_Float16)U1[k * 1024 + n];
  }
  for (int i = i0; i < 1024 * 512; i += stride) {         // [W2;U2] -> [1024][512]
    int n = i >> 9, k = i & 511;
    bp2[i] = (_Float16)(k < 256 ? W2[k * 1024 + n] : U2[(k - 256) * 1024 + n]);
  }
  for (int i = i0; i < T_ * RP * 32; i += stride) {       // x -> [T][208][32] f16
    int t = i / (RP * 32);
    int rem = i - t * (RP * 32);
    int r = rem >> 5, f = rem & 31;
    float v = (r < B_ && f < F_) ? x[((size_t)r * T_ + t) * F_ + f] : 0.0f;
    xp[i] = (_Float16)v;
  }
}

// ---------------- persistent double-LSTM kernel ----------------
__global__ __launch_bounds__(NTHR) void k_lstm(
    const float* __restrict__ b1, const float* __restrict__ b2,
    const _Float16* __restrict__ bp0, const _Float16* __restrict__ bp1,
    const _Float16* __restrict__ bp2, const _Float16* __restrict__ xp,
    _Float16* __restrict__ hbase,     /* [4][208][256] f16: h1 x2, h2 x2 */
    float* __restrict__ c1, float* __restrict__ c2,
    float* __restrict__ h2f, int* __restrict__ bar) {
  extern __shared__ _Float16 lds[];

  const int ct = blockIdx.x;            // 0..15 : H-column tile of this block
  const int ct0 = ct * 16;
  const int w  = threadIdx.x >> 5;      // 0..12 : row tile
  const int lane = threadIdx.x & 31;
  const int r0 = w * 16;
  const int nlo = lane & 15;
  const int kh  = lane >> 4;
  const int arow = r0 + nlo;

  // ------- stage this block's weight slab into LDS (once) -------
  for (int i = threadIdx.x; i < 4 * 16 * 32; i += NTHR) {        // W1p
    int g = i >> 9, nl = (i >> 5) & 15, k = i & 31;
    lds[LB0_OFF + i] = bp0[(size_t)(g * 256 + ct0 + nl) * 32 + k];
  }
  for (int i = threadIdx.x; i < 4 * 16 * 256; i += NTHR) {       // U1
    int g = i >> 12, nl = (i >> 8) & 15, k = i & 255;
    lds[LB1_OFF + i] = bp1[(size_t)(g * 256 + ct0 + nl) * 256 + k];
  }
  for (int i = threadIdx.x; i < 4 * 16 * 512; i += NTHR) {       // [W2;U2]
    int g = i >> 13, nl = (i >> 9) & 15, k = i & 511;
    lds[LB2_OFF + i] = bp2[(size_t)(g * 256 + ct0 + nl) * 512 + k];
  }
  __syncthreads();

  // per-lane base indices into the LDS slabs (gate x local-column)
  const int kofs = kh * 16;
  const int i0I = LB0_OFF + (0 * 16 + nlo) * 32 + kofs;
  const int i0F = LB0_OFF + (1 * 16 + nlo) * 32 + kofs;
  const int i0G = LB0_OFF + (2 * 16 + nlo) * 32 + kofs;
  const int i0O = LB0_OFF + (3 * 16 + nlo) * 32 + kofs;
  const int i1I = LB1_OFF + (0 * 16 + nlo) * 256 + kofs;
  const int i1F = LB1_OFF + (1 * 16 + nlo) * 256 + kofs;
  const int i1G = LB1_OFF + (2 * 16 + nlo) * 256 + kofs;
  const int i1O = LB1_OFF + (3 * 16 + nlo) * 256 + kofs;
  const int i2I = LB2_OFF + (0 * 16 + nlo) * 512 + kofs;
  const int i2F = LB2_OFF + (1 * 16 + nlo) * 512 + kofs;
  const int i2G = LB2_OFF + (2 * 16 + nlo) * 512 + kofs;
  const int i2O = LB2_OFF + (3 * 16 + nlo) * 512 + kofs;

  const int ncol = ct0 + nlo;
  const float bi1 = b1[ncol], bf1 = b1[256 + ncol], bg1 = b1[512 + ncol], bo1 = b1[768 + ncol];
  const float bi2 = b2[ncol], bf2 = b2[256 + ncol], bg2 = b2[512 + ncol], bo2 = b2[768 + ncol];

  // C/D layout: vgpr r -> M = r + 8*kh, N = nlo
  const int cbase = (r0 + (kh << 3)) * H_ + ct0 + nlo;

  for (int t = 0; t < T_; ++t) {
    const int p = t & 1;
    const int h1r_off = p * RPH;
    const int h1w_off = (1 - p) * RPH;
    const int h2r_off = (2 + p) * RPH;
    const int h2w_off = (3 - p) * RPH;

    // ---------- phase A: layer-1  z1 = x_t@W1p + h1@U1 + b1 ----------
    const int za = opaque_zero();     // loop-variant 0: blocks LICM of ds loads
    v8f zi = {bi1, bi1, bi1, bi1, bi1, bi1, bi1, bi1};
    v8f zf = {bf1, bf1, bf1, bf1, bf1, bf1, bf1, bf1};
    v8f zg = {bg1, bg1, bg1, bg1, bg1, bg1, bg1, bg1};
    v8f zo = {bo1, bo1, bo1, bo1, bo1, bo1, bo1, bo1};
    {
      v16h a = load_a(xp + (size_t)t * RP * 32, 32, arow, 0, kh);
      zi = wmma_f16(a, lds_b(lds, i0I + za), zi);
      zf = wmma_f16(a, lds_b(lds, i0F + za), zf);
      zg = wmma_f16(a, lds_b(lds, i0G + za), zg);
      zo = wmma_f16(a, lds_b(lds, i0O + za), zo);
    }
#pragma unroll
    for (int kb = 0; kb < H_; kb += 32) {
      v16h a = load_a(hbase + h1r_off, H_, arow, kb, kh);
      zi = wmma_f16(a, lds_b(lds, i1I + kb + za), zi);
      zf = wmma_f16(a, lds_b(lds, i1F + kb + za), zf);
      zg = wmma_f16(a, lds_b(lds, i1G + kb + za), zg);
      zo = wmma_f16(a, lds_b(lds, i1O + kb + za), zo);
    }
#pragma unroll
    for (int r = 0; r < 8; ++r) {
      int off = cbase + r * H_;
      float iv = sigf(zi[r]), fv = sigf(zf[r]);
      float gv = tanhf(zg[r]), ov = sigf(zo[r]);
      float cn = fv * c1[off] + iv * gv;
      c1[off] = cn;
      hbase[h1w_off + off] = (_Float16)(ov * tanhf(cn));
    }
    grid_barrier(bar);

    // ---------- phase B: layer-2  z2 = [h1_t | h2_{t-1}] @ [W2;U2] + b2 ----------
    const int zb = opaque_zero();
    zi = (v8f){bi2, bi2, bi2, bi2, bi2, bi2, bi2, bi2};
    zf = (v8f){bf2, bf2, bf2, bf2, bf2, bf2, bf2, bf2};
    zg = (v8f){bg2, bg2, bg2, bg2, bg2, bg2, bg2, bg2};
    zo = (v8f){bo2, bo2, bo2, bo2, bo2, bo2, bo2, bo2};
#pragma unroll
    for (int kb = 0; kb < 512; kb += 32) {
      const int aoff = (kb < 256) ? h1w_off : h2r_off;
      const int kc = kb & 255;
      v16h a = load_a(hbase + aoff, H_, arow, kc, kh);
      zi = wmma_f16(a, lds_b(lds, i2I + kb + zb), zi);
      zf = wmma_f16(a, lds_b(lds, i2F + kb + zb), zf);
      zg = wmma_f16(a, lds_b(lds, i2G + kb + zb), zg);
      zo = wmma_f16(a, lds_b(lds, i2O + kb + zb), zo);
    }
#pragma unroll
    for (int r = 0; r < 8; ++r) {
      int off = cbase + r * H_;
      float iv = sigf(zi[r]), fv = sigf(zf[r]);
      float gv = tanhf(zg[r]), ov = sigf(zo[r]);
      float cn = fv * c2[off] + iv * gv;
      c2[off] = cn;
      float hv = ov * tanhf(cn);
      hbase[h2w_off + off] = (_Float16)hv;
      h2f[off] = hv;                 // f32 copy; last step feeds the head
    }
    grid_barrier(bar);
  }
}

// ---------------- dense(relu) + softmax head ----------------
__global__ void k_head(const float* __restrict__ h2f, const float* __restrict__ Wd,
                       const float* __restrict__ bd, const float* __restrict__ Ws,
                       const float* __restrict__ bs, float* __restrict__ out) {
  __shared__ float hid[D_];
  __shared__ float logit[C_];
  const int b = blockIdx.x;
  const float* h = h2f + (size_t)b * H_;
  for (int d = threadIdx.x; d < D_; d += blockDim.x) {
    float s = bd[d];
    for (int k = 0; k < H_; ++k) s += h[k] * Wd[k * D_ + d];
    hid[d] = fmaxf(s, 0.0f);
  }
  __syncthreads();
  if (threadIdx.x < C_) {
    int c = threadIdx.x;
    float s = bs[c];
    for (int d = 0; d < D_; ++d) s += hid[d] * Ws[d * C_ + c];
    logit[c] = s;
  }
  __syncthreads();
  if (threadIdx.x == 0) {
    float m = logit[0];
    for (int c = 1; c < C_; ++c) m = fmaxf(m, logit[c]);
    float e[C_], sum = 0.0f;
    for (int c = 0; c < C_; ++c) { e[c] = expf(logit[c] - m); sum += e[c]; }
    float inv = 1.0f / sum;
    for (int c = 0; c < C_; ++c) out[(size_t)b * C_ + c] = e[c] * inv;
  }
}

// ---------------- host launcher ----------------
extern "C" void kernel_launch(void* const* d_in, const int* in_sizes, int n_in,
                              void* d_out, int out_size, void* d_ws, size_t ws_size,
                              hipStream_t stream) {
  (void)in_sizes; (void)n_in; (void)out_size; (void)ws_size;
  const float* x  = (const float*)d_in[0];
  const float* W1 = (const float*)d_in[1];
  const float* U1 = (const float*)d_in[2];
  const float* b1 = (const float*)d_in[3];
  const float* W2 = (const float*)d_in[4];
  const float* U2 = (const float*)d_in[5];
  const float* b2 = (const float*)d_in[6];
  const float* Wd = (const float*)d_in[7];
  const float* bd = (const float*)d_in[8];
  const float* Ws = (const float*)d_in[9];
  const float* bs = (const float*)d_in[10];

  char* ws = (char*)d_ws;
  int*       bar   = (int*)(ws + OFF_BAR);
  _Float16*  bp0   = (_Float16*)(ws + OFF_BP0);
  _Float16*  bp1   = (_Float16*)(ws + OFF_BP1);
  _Float16*  bp2   = (_Float16*)(ws + OFF_BP2);
  _Float16*  xp    = (_Float16*)(ws + OFF_XP);
  _Float16*  hbase = (_Float16*)(ws + OFF_H1);
  float*     c1    = (float*)(ws + OFF_C1);
  float*     c2    = (float*)(ws + OFF_C2);
  float*     h2f   = (float*)(ws + OFF_H2F);

  const int nwords = (int)((OFF_END - OFF_H1) / 4);
  k_init<<<256, 256, 0, stream>>>(bar, (float*)(ws + OFF_H1), nwords);
  k_prep<<<1024, 256, 0, stream>>>(W1, U1, W2, U2, x, bp0, bp1, bp2, xp);
  k_lstm<<<NBLK, NTHR, LDS_BYTES, stream>>>(b1, b2, bp0, bp1, bp2, xp,
                                            hbase, c1, c2, h2f, bar);
  k_head<<<B_, 192, 0, stream>>>(h2f, Wd, bd, Ws, bs, (float*)d_out);
}
